// NearestNeighborAttention_30202210025911
// MI455X (gfx1250) — matrix operand order repaired and checked by reading, not compiled, not run on hardware
//
#include <hip/hip_runtime.h>
#include <hip/hip_bf16.h>
#include <float.h>

#define B_   2
#define S_   4096
#define F_   1024
#define H_   16
#define HD_  64
#define KNB_ 16

typedef __attribute__((ext_vector_type(16))) __bf16 v16bf;
typedef __attribute__((ext_vector_type(8)))  float  v8f;
typedef int v4i __attribute__((vector_size(16)));

__device__ __forceinline__ unsigned short f2bf(float f) {
    unsigned int u = __float_as_uint(f);
    u += 0x7FFFu + ((u >> 16) & 1u);          // round-to-nearest-even
    return (unsigned short)(u >> 16);
}

// CDNA5 async global->LDS copy (16B per lane), tracked by ASYNCcnt.
__device__ __forceinline__ void async_cp16(const unsigned short* g, unsigned short* l) {
    __builtin_amdgcn_global_load_async_to_lds_b128(
        (__attribute__((address_space(1))) v4i*)(g),
        (__attribute__((address_space(3))) v4i*)(l),
        0, 0);
}

// ---------------------------------------------------------------- fp32 -> bf16
__global__ void cvt_bf16_kernel(const float* __restrict__ in,
                                unsigned short* __restrict__ out, int n) {
    int i = (blockIdx.x * blockDim.x + threadIdx.x) * 4;
    if (i + 3 < n) {
        float4 f = *(const float4*)(in + i);
        unsigned int lo = (unsigned)f2bf(f.x) | ((unsigned)f2bf(f.y) << 16);
        unsigned int hi = (unsigned)f2bf(f.z) | ((unsigned)f2bf(f.w) << 16);
        *(uint2*)(out + i) = make_uint2(lo, hi);
    }
}

// ---------------------------------------------------------------- KNN (top-16)
__global__ __launch_bounds__(256)
void knn_kernel(const float* __restrict__ coords, int* __restrict__ nn_idx) {
    __shared__ float sc[S_ * 3];                       // 48 KB
    const int b = blockIdx.y;
    const float* cb = coords + (size_t)b * S_ * 3;
    for (int i = threadIdx.x; i < S_ * 3; i += blockDim.x) sc[i] = cb[i];
    __syncthreads();

    const int q = blockIdx.x * blockDim.x + threadIdx.x;
    const float qx = sc[q*3+0], qy = sc[q*3+1], qz = sc[q*3+2];

    float bd[KNB_]; int bi[KNB_];
#pragma unroll
    for (int i = 0; i < KNB_; ++i) { bd[i] = FLT_MAX; bi[i] = -1; }

    for (int j = 0; j < S_; ++j) {
        float dx = sc[j*3+0] - qx, dy = sc[j*3+1] - qy, dz = sc[j*3+2] - qz;
        float d2 = dx*dx + dy*dy + dz*dz;
        if (j == q) continue;
        if (d2 < bd[KNB_-1]) {
            float dc = d2; int ic = j;
#pragma unroll
            for (int i = 0; i < KNB_; ++i) {           // register insertion pass
                if (dc < bd[i]) {
                    float tf = bd[i]; bd[i] = dc; dc = tf;
                    int   ti = bi[i]; bi[i] = ic; ic = ti;
                }
            }
        }
    }
    int* o = nn_idx + ((size_t)b * S_ + q) * KNB_;
#pragma unroll
    for (int i = 0; i < KNB_; ++i) o[i] = bi[i];
}

// ------------------------------------------------- y = x @ W^T  (bf16 WMMA)
// M = 8192, N = K = 1024. Block: 8 waves = 4(M) x 2(N); output tile 128x128.
// Double-buffered LDS staging via GLOBAL_LOAD_ASYNC_TO_LDS_B128 (ASYNCcnt).
#define KT_   32            // K per stage
#define APAD_ 40            // padded row length (elements) -> 80B rows
__global__ __launch_bounds__(256)
void gemm_wmma_kernel(const unsigned short* __restrict__ xb,   // [8192,1024] bf16
                      const unsigned short* __restrict__ wb,   // [1024,1024] bf16
                      float* __restrict__ out) {               // [8192,1024] f32
    __shared__ unsigned short As[2][128][APAD_];   // 2 x 10 KB
    __shared__ unsigned short Bs[2][128][APAD_];   // 2 x 10 KB

    const int tid  = threadIdx.x;
    const int lane = tid & 31;
    const int wave = tid >> 5;
    const int half = lane >> 4;          // K-half selector per ISA layout
    const int lrow = lane & 15;          // M (A) / N (B,D) index
    const int wm   = wave & 3;           // wave M offset = wm*32
    const int wn   = wave >> 2;          // wave N offset = wn*64
    const int mBlock = blockIdx.x * 128;
    const int nBlock = blockIdx.y * 128;

    // staging: 512 chunks of 16B per tile; thread owns chunks tid and tid+256
    const int r0 = tid >> 2,         k0 = (tid & 3) * 8;
    const int r1 = (tid + 256) >> 2, k1 = ((tid + 256) & 3) * 8;

    v8f acc[2][4] = {};

    auto stage = [&](int buf, int kb) {
        async_cp16(xb + (size_t)(mBlock + r0) * F_ + kb + k0, &As[buf][r0][k0]);
        async_cp16(xb + (size_t)(mBlock + r1) * F_ + kb + k1, &As[buf][r1][k1]);
        async_cp16(wb + (size_t)(nBlock + r0) * F_ + kb + k0, &Bs[buf][r0][k0]);
        async_cp16(wb + (size_t)(nBlock + r1) * F_ + kb + k1, &Bs[buf][r1][k1]);
    };

    stage(0, 0);
#pragma clang loop unroll(disable)
    for (int kt = 0; kt < F_ / KT_; ++kt) {
        const int cur = kt & 1;
        if (kt + 1 < F_ / KT_) {
            stage(cur ^ 1, (kt + 1) * KT_);
            asm volatile("s_wait_asynccnt 0x4" ::: "memory");  // cur buffer landed
        } else {
            asm volatile("s_wait_asynccnt 0x0" ::: "memory");
        }
        __syncthreads();

        // Both A fragments for this wave's 32 M-rows
        union { v16bf v; uint4 q[2]; } Af0, Af1;
        const int rA = wm * 32 + lrow;
        Af0.q[0] = *(const uint4*)&As[cur][rA][half * 8];
        Af0.q[1] = *(const uint4*)&As[cur][rA][16 + half * 8];
        Af1.q[0] = *(const uint4*)&As[cur][rA + 16][half * 8];
        Af1.q[1] = *(const uint4*)&As[cur][rA + 16][16 + half * 8];

        // Stream one B fragment at a time; 2 WMMAs per fragment
#pragma unroll
        for (int j = 0; j < 4; ++j) {
            union { v16bf v; uint4 q[2]; } Bf;
            const int rB = wn * 64 + j * 16 + lrow;
            Bf.q[0] = *(const uint4*)&Bs[cur][rB][half * 8];
            Bf.q[1] = *(const uint4*)&Bs[cur][rB][16 + half * 8];
            acc[0][j] = __builtin_amdgcn_wmma_f32_16x16x32_bf16(
                false, Af0.v, false, Bf.v, (short)0, acc[0][j], false, false);
            acc[1][j] = __builtin_amdgcn_wmma_f32_16x16x32_bf16(
                false, Af1.v, false, Bf.v, (short)0, acc[1][j], false, false);
        }
        __syncthreads();   // buffer 'cur' free for restage next iteration
    }

    // D layout: VGPR r holds M = r + 8*half, N = lrow
#pragma unroll
    for (int i = 0; i < 2; ++i) {
#pragma unroll
        for (int j = 0; j < 4; ++j) {
            float* ob = out + (size_t)(mBlock + wm * 32 + i * 16) * F_
                            + nBlock + wn * 64 + j * 16 + lrow;
#pragma unroll
            for (int r = 0; r < 8; ++r)
                ob[(size_t)(r + half * 8) * F_] = acc[i][j][r];
        }
    }
}

// ------------------------------------------------- sparse neighbor attention
__global__ __launch_bounds__(256)
void attn_kernel(const float* __restrict__ Q, const float* __restrict__ K,
                 const float* __restrict__ V, const int* __restrict__ nn_idx,
                 float* __restrict__ out) {
    const int wid  = blockIdx.x * 8 + (threadIdx.x >> 5);   // one wave per (b,s,h)
    const int lane = threadIdx.x & 31;
    const int h = wid & (H_ - 1);
    const int s = (wid >> 4) & (S_ - 1);
    const int b = wid >> 16;

    const size_t rowq = ((size_t)(b * S_ + s)) * F_ + h * HD_;
    const float2 qv = *(const float2*)(Q + rowq + lane * 2);
    const int* nb = nn_idx + ((size_t)(b * S_ + s)) * KNB_;

    float sc[KNB_];
#pragma unroll
    for (int j = 0; j < KNB_; ++j) {
        const int idx = nb[j];
        const float2 kv =
            *(const float2*)(K + ((size_t)(b * S_ + idx)) * F_ + h * HD_ + lane * 2);
        float p = qv.x * kv.x + qv.y * kv.y;
#pragma unroll
        for (int o = 16; o > 0; o >>= 1) p += __shfl_xor(p, o, 32);
        sc[j] = p * 0.125f;                               // 1/sqrt(64)
    }
    float m = sc[0];
#pragma unroll
    for (int j = 1; j < KNB_; ++j) m = fmaxf(m, sc[j]);
    float sum = 0.f;
#pragma unroll
    for (int j = 0; j < KNB_; ++j) { sc[j] = __expf(sc[j] - m); sum += sc[j]; }
    const float inv = 1.0f / sum;

    float ox = 0.f, oy = 0.f;
#pragma unroll
    for (int j = 0; j < KNB_; ++j) {
        const int idx = nb[j];
        const float2 vv =
            *(const float2*)(V + ((size_t)(b * S_ + idx)) * F_ + h * HD_ + lane * 2);
        const float w = sc[j] * inv;
        ox += w * vv.x; oy += w * vv.y;
    }
    *(float2*)(out + rowq + lane * 2) = make_float2(ox, oy);
}

// ------------------------------------------------- metric = K.mean over heads
__global__ void metric_kernel(const float* __restrict__ K, float* __restrict__ metric) {
    const int t = blockIdx.x * blockDim.x + threadIdx.x;   // 0..B*S*HD-1
    const int d  = t & (HD_ - 1);
    const int sg = t >> 6;
    const float* kb = K + (size_t)sg * F_ + d;
    float s = 0.f;
#pragma unroll
    for (int h = 0; h < H_; ++h) s += kb[h * HD_];
    metric[t] = s * (1.0f / 16.0f);
}

// ------------------------------------------------------------------ launcher
extern "C" void kernel_launch(void* const* d_in, const int* in_sizes, int n_in,
                              void* d_out, int out_size, void* d_ws, size_t ws_size,
                              hipStream_t stream) {
    const float* x      = (const float*)d_in[0];
    const float* coords = (const float*)d_in[1];
    const float* Wq     = (const float*)d_in[2];
    const float* Wk     = (const float*)d_in[3];
    const float* Wv     = (const float*)d_in[4];
    float* out = (float*)d_out;

    char* ws = (char*)d_ws;
    float*          Qb  = (float*)(ws + 0);
    float*          Kb  = (float*)(ws + 33554432);
    float*          Vb  = (float*)(ws + 67108864);
    unsigned short* xb  = (unsigned short*)(ws + 100663296);
    unsigned short* wqb = (unsigned short*)(ws + 117440512);
    unsigned short* wkb = wqb + 1048576;
    unsigned short* wvb = wkb + 1048576;
    int*            nn  = (int*)(ws + 123731968);

    const int NX = B_ * S_ * F_;        // 8388608
    const int NW = F_ * F_;             // 1048576

    cvt_bf16_kernel<<<NX / 4 / 256, 256, 0, stream>>>(x,  xb,  NX);
    cvt_bf16_kernel<<<NW / 4 / 256, 256, 0, stream>>>(Wq, wqb, NW);
    cvt_bf16_kernel<<<NW / 4 / 256, 256, 0, stream>>>(Wk, wkb, NW);
    cvt_bf16_kernel<<<NW / 4 / 256, 256, 0, stream>>>(Wv, wvb, NW);

    knn_kernel<<<dim3(S_ / 256, B_), 256, 0, stream>>>(coords, nn);

    dim3 ggrid((B_ * S_) / 128, F_ / 128);
    gemm_wmma_kernel<<<ggrid, 256, 0, stream>>>(xb, wqb, Qb);
    gemm_wmma_kernel<<<ggrid, 256, 0, stream>>>(xb, wkb, Kb);
    gemm_wmma_kernel<<<ggrid, 256, 0, stream>>>(xb, wvb, Vb);

    attn_kernel<<<(B_ * S_ * H_) / 8, 256, 0, stream>>>(Qb, Kb, Vb, nn, out);

    metric_kernel<<<(B_ * S_ * HD_) / 256, 256, 0, stream>>>(Kb, out + (size_t)NX);
}
